// multihead_self_attention_70377334112366
// MI455X (gfx1250) — matrix operand order repaired and checked by reading, not compiled, not run on hardware
//
#include <hip/hip_runtime.h>

// ---------------------------------------------------------------------------
// Multi-head self-attention with RoPE for MI455X (gfx1250, wave32, WMMA).
//   x:[8,1024,2048] f32, wq/wk/wv/wo:[2048,2048] f32 -> out:[8,1024,2048] f32
// Pipeline: bf16 convert -> fused QKV GEMM (+RoPE epilogue, double-buffered
// TDM-staged A tile) -> flash attention (bf16 WMMA, online softmax)
// -> output GEMM.
// ---------------------------------------------------------------------------

typedef __attribute__((ext_vector_type(16))) __bf16 v16bf;
typedef __attribute__((ext_vector_type(8)))  __bf16 v8bf;
typedef __attribute__((ext_vector_type(8)))  float  v8f;
typedef __attribute__((ext_vector_type(4)))  unsigned int v4u;
typedef __attribute__((ext_vector_type(8)))  int v8i;
typedef __attribute__((ext_vector_type(4)))  int v4i;

static constexpr int B_    = 8;
static constexpr int S_    = 1024;
static constexpr int D_    = 2048;
static constexpr int H_    = 16;
static constexpr int DK    = 128;
static constexpr int BS    = B_ * S_;     // 8192 rows
static constexpr int NQKV  = 3 * D_;      // 6144
static constexpr int THREADS = 256;       // 8 waves (wave32)
static constexpr int TILE_ELEMS = 128 * 32;

#if defined(__HIP_DEVICE_COMPILE__) && __has_builtin(__builtin_amdgcn_tensor_load_to_lds)
#define USE_TDM 1
#else
#define USE_TDM 0
#endif

// ---------------------------------------------------------------- fragments

__device__ inline v16bf frag_combine(v8bf lo, v8bf hi) {
    v16bf r;
#pragma unroll
    for (int i = 0; i < 8; ++i) { r[i] = lo[i]; r[i + 8] = hi[i]; }
    return r;
}

// A fragment 16x32 bf16 from row-major src (ldr elements per row).
// lanes 0-15: M=lane, K 0-7 (VGPR0-3) and 16-23 (VGPR4-7)
// lanes 16-31: same M, K 8-15 and 24-31.
__device__ inline v16bf load_a_frag(const __bf16* src, int ldr) {
    int lane = threadIdx.x & 31;
    int row  = lane & 15;
    int k0   = (lane < 16) ? 0 : 8;
    v8bf lo = *reinterpret_cast<const v8bf*>(src + (size_t)row * ldr + k0);
    v8bf hi = *reinterpret_cast<const v8bf*>(src + (size_t)row * ldr + k0 + 16);
    return frag_combine(lo, hi);
}

// B fragment 32x16 bf16, B[k][n] = src[n*ldr + k] (weight-style [N,K] rows).
// lanes 0-15: N=lane, K 0-7 and 16-23; lanes 16-31: same N, K 8-15 and 24-31.
__device__ inline v16bf load_b_frag(const __bf16* src, int ldr) {
    int lane = threadIdx.x & 31;
    int row  = lane & 15;
    int k0   = (lane < 16) ? 0 : 8;
    v8bf lo = *reinterpret_cast<const v8bf*>(src + (size_t)row * ldr + k0);
    v8bf hi = *reinterpret_cast<const v8bf*>(src + (size_t)row * ldr + k0 + 16);
    return frag_combine(lo, hi);
}

__device__ inline v8f wmma_bf16(v16bf a, v16bf b, v8f c) {
    return __builtin_amdgcn_wmma_f32_16x16x32_bf16(
        false, a, false, b, (short)0, c, false, false);
}

__device__ inline v8f zero8() {
    v8f z = {0.f, 0.f, 0.f, 0.f, 0.f, 0.f, 0.f, 0.f};
    return z;
}

// ------------------------------------------------------------------- TDM

#if USE_TDM
// 2D tile (tile_rows x tile_cols elements, 2B each) global -> LDS (packed
// row-major, row stride = tile_cols). Descriptor per CDNA5 ISA ch. 8.
__device__ inline void tdm_load_tile_bf16(unsigned lds_off, const void* gaddr,
                                          unsigned tile_cols, unsigned tile_rows,
                                          unsigned tensor_cols, unsigned tensor_rows,
                                          unsigned row_stride_elems) {
    unsigned long long ga = (unsigned long long)gaddr;
    v4u g0;
    g0[0] = 1u;                                   // count=1, user descriptor
    g0[1] = lds_off;                              // lds_addr
    g0[2] = (unsigned)ga;                         // global_addr[31:0]
    g0[3] = (unsigned)((ga >> 32) & 0x1FFFFFFull) | (2u << 30);  // addr[56:32], type=2
    v8i g1;
    g1[0] = (int)(1u << 16);                      // wg_mask=0, data_size=1 (2 bytes)
    g1[1] = (int)((tensor_cols & 0xFFFFu) << 16); // tensor_dim0[15:0]
    g1[2] = (int)(((tensor_cols >> 16) & 0xFFFFu) | ((tensor_rows & 0xFFFFu) << 16));
    g1[3] = (int)(((tensor_rows >> 16) & 0xFFFFu) | ((tile_cols & 0xFFFFu) << 16));
    g1[4] = (int)(tile_rows & 0xFFFFu);           // tile_dim1, tile_dim2=0
    g1[5] = (int)row_stride_elems;                // tensor_dim0_stride[31:0]
    g1[6] = 0;
    g1[7] = 0;
    v4i z4 = {0, 0, 0, 0};
#if __clang_major__ >= 23
    v8i z8 = {0, 0, 0, 0, 0, 0, 0, 0};
    __builtin_amdgcn_tensor_load_to_lds(g0, g1, z4, z4, z8, 0);
#else
    __builtin_amdgcn_tensor_load_to_lds(g0, g1, z4, z4, 0);
#endif
}
#endif

// ------------------------------------------------------------- convert

__global__ void cvt_f32_bf16_kernel(const float* __restrict__ src,
                                    __bf16* __restrict__ dst, int n) {
    int i = blockIdx.x * blockDim.x + threadIdx.x;
    int stride = gridDim.x * blockDim.x;
    for (; i < n; i += stride) dst[i] = (__bf16)src[i];
}

// ------------------------------------------------- shared GEMM mainloop
// Block tile 128(M) x 128(N), K-step 32, 8 waves in 2(M) x 4(N) grid.
// Wave computes 64x32 = 4x2 fragments. A tile double-buffered in LDS via
// TDM (one DMA in flight behind the WMMAs); B fragments read straight from
// L2-resident weights.

__device__ inline void gemm_mainloop(const __bf16* __restrict__ X,
                                     const __bf16* __restrict__ W,
                                     int Kdim, int Mtot,
                                     __bf16* smA /* 2 * TILE_ELEMS */,
                                     v8f acc[4][2]) {
    const int wave = threadIdx.x >> 5;
    const int wm = wave & 1;        // 0..1
    const int wn = wave >> 1;       // 0..3
    const int bm = blockIdx.x;
    const int bn = blockIdx.y;
    const int nkt = Kdim / 32;

#if USE_TDM
    // Prologue: stage K-tile 0 into buffer 0.
    if (threadIdx.x < 32) {
        tdm_load_tile_bf16((unsigned)(size_t)smA,
                           X + (size_t)(bm * 128) * Kdim,
                           /*tile_cols=*/32, /*tile_rows=*/128,
                           /*tensor_cols=*/(unsigned)Kdim,
                           /*tensor_rows=*/(unsigned)(Mtot - bm * 128),
                           /*row_stride=*/(unsigned)Kdim);
    }
#endif

    for (int kt = 0; kt < nkt; ++kt) {
        const int k = kt * 32;
        __bf16* bufc = smA + (kt & 1) * TILE_ELEMS;
#if USE_TDM
        // Current buffer's DMA has landed; previous buffer's readers are done
        // (each wave's ds_loads complete before its consuming WMMA, which
        // precedes its barrier signal) -> safe to launch the next DMA.
        if (threadIdx.x < 32)
            __builtin_amdgcn_s_wait_tensorcnt(0);
        __syncthreads();
        if (kt + 1 < nkt && threadIdx.x < 32) {
            __bf16* bufn = smA + ((kt + 1) & 1) * TILE_ELEMS;
            tdm_load_tile_bf16((unsigned)(size_t)bufn,
                               X + (size_t)(bm * 128) * Kdim + (k + 32),
                               32, 128,
                               (unsigned)(Kdim - (k + 32)),
                               (unsigned)(Mtot - bm * 128),
                               (unsigned)Kdim);
        }
#else
        __syncthreads();
        for (int t = threadIdx.x; t < TILE_ELEMS / 8; t += THREADS) {
            int row = t >> 2, c = t & 3;
            reinterpret_cast<v8bf*>(bufc)[t] =
                *reinterpret_cast<const v8bf*>(X + (size_t)(bm * 128 + row) * Kdim + k + c * 8);
        }
        __syncthreads();
#endif

        v16bf a[4], bf[2];
#pragma unroll
        for (int i = 0; i < 4; ++i)
            a[i] = load_a_frag(bufc + (wm * 64 + i * 16) * 32, 32);
#pragma unroll
        for (int j = 0; j < 2; ++j) {
            const __bf16* wrow = W + (size_t)(bn * 128 + wn * 32 + j * 16) * Kdim + k;
            bf[j] = load_b_frag(wrow, Kdim);
            if (kt + 1 < nkt)
                __builtin_prefetch(wrow + 32, 0, 3);   // global_prefetch_b8
        }
#pragma unroll
        for (int i = 0; i < 4; ++i)
#pragma unroll
            for (int j = 0; j < 2; ++j)
                acc[i][j] = wmma_bf16(a[i], bf[j], acc[i][j]);
    }
}

// ------------------------------------------- QKV projection + RoPE epilogue
// y[m,n] = sum_i X[m,i]*Wqkv[n,i]; n<2048 -> Q, <4096 -> K, else -> V.
// RoPE pairs (2p,2p+1) are adjacent lanes of the C fragment -> shfl_xor(1).

__global__ __launch_bounds__(THREADS)
void gemm_qkv_rope_kernel(const __bf16* __restrict__ X,
                          const __bf16* __restrict__ W,
                          __bf16* __restrict__ Qb,
                          __bf16* __restrict__ Kb,
                          __bf16* __restrict__ Vt) {
    __shared__ __bf16 smA[2 * TILE_ELEMS];
    v8f acc[4][2];
#pragma unroll
    for (int i = 0; i < 4; ++i)
#pragma unroll
        for (int j = 0; j < 2; ++j) acc[i][j] = zero8();

    gemm_mainloop(X, W, D_, BS, smA, acc);

    const int lane = threadIdx.x & 31;
    const int wave = threadIdx.x >> 5;
    const int wm = wave & 1, wn = wave >> 1;
    const int half = (lane >> 4) & 1;
    const float LN1E4_OVER_64 = 0.14391156731f;  // ln(10000)/64

#pragma unroll
    for (int i = 0; i < 4; ++i) {
#pragma unroll
        for (int j = 0; j < 2; ++j) {
            const int n = blockIdx.y * 128 + wn * 32 + j * 16 + (lane & 15);
            const int mbase = blockIdx.x * 128 + wm * 64 + i * 16 + half * 8;
#pragma unroll
            for (int r = 0; r < 8; ++r) {
                const int m = mbase + r;
                const int b = m >> 10, s = m & (S_ - 1);
                float val = acc[i][j][r];
                float partner = __shfl_xor(val, 1, 32);   // RoPE pair element
                if (n < 2 * D_) {
                    int p = (n & (DK - 1)) >> 1;
                    float inv = __expf(-(float)p * LN1E4_OVER_64);
                    float ang = (float)s * inv;
                    float sn, cs;
                    __sincosf(ang, &sn, &cs);
                    float res = (n & 1) ? (val * cs + partner * sn)
                                        : (val * cs - partner * sn);
                    if (n < D_) {
                        int h = n >> 7;
                        Qb[((size_t)(b * H_ + h) * S_ + s) * DK + (n & (DK - 1))] = (__bf16)res;
                    } else {
                        int h = (n - D_) >> 7;
                        Kb[((size_t)(b * H_ + h) * S_ + s) * DK + (n & (DK - 1))] = (__bf16)res;
                    }
                } else {
                    int h = (n - 2 * D_) >> 7;   // V stored transposed [B,H,dk,S]
                    Vt[((size_t)(b * H_ + h) * DK + (n & (DK - 1))) * S_ + s] = (__bf16)val;
                }
            }
        }
    }
}

// ------------------------------------------------------- flash attention
// Block = (b,h, 128-query tile); 8 waves x 16 query rows. Online softmax
// over 32-key tiles; P staged via per-wave LDS to transpose C- -> A-layout.

__device__ inline float rowmax16(float v) {
#pragma unroll
    for (int m = 1; m < 16; m <<= 1) v = fmaxf(v, __shfl_xor(v, m, 32));
    return v;
}
__device__ inline float rowsum16(float v) {
#pragma unroll
    for (int m = 1; m < 16; m <<= 1) v += __shfl_xor(v, m, 32);
    return v;
}

__global__ __launch_bounds__(THREADS)
void flash_attn_kernel(const __bf16* __restrict__ Qb,
                       const __bf16* __restrict__ Kb,
                       const __bf16* __restrict__ Vt,
                       __bf16* __restrict__ O) {
    __shared__ __bf16 smP[8][16 * 32];   // 1KB per wave
    const int lane = threadIdx.x & 31;
    const int w    = threadIdx.x >> 5;
    const int half = (lane >> 4) & 1;
    const int qt = blockIdx.x & 7;
    const int bh = blockIdx.x >> 3;      // b*16 + h
    const int q0 = qt * 128 + w * 16;
    const float SCALE = 0.08838834764831845f;   // 1/sqrt(128)

    const __bf16* Qp = Qb + ((size_t)bh * S_ + q0) * DK;
    const __bf16* Kp = Kb + (size_t)bh * S_ * DK;
    const __bf16* Vp = Vt + (size_t)bh * DK * S_;

    v16bf qa[4];
#pragma unroll
    for (int kk = 0; kk < 4; ++kk) qa[kk] = load_a_frag(Qp + kk * 32, DK);

    v8f o[8];
    float mrow[8], lrow[8];
#pragma unroll
    for (int nb = 0; nb < 8; ++nb) o[nb] = zero8();
#pragma unroll
    for (int r = 0; r < 8; ++r) { mrow[r] = -1e30f; lrow[r] = 0.f; }

    const int ktend = (qt * 128 + 127) >> 5;   // causal: keys <= block max query
    for (int kt = 0; kt <= ktend; ++kt) {
        const int kb = kt * 32;
        v8f s0 = zero8(), s1 = zero8();
#pragma unroll
        for (int kk = 0; kk < 4; ++kk) {
            v16bf b0 = load_b_frag(Kp + (size_t)kb * DK + kk * 32, DK);
            v16bf b1 = load_b_frag(Kp + (size_t)(kb + 16) * DK + kk * 32, DK);
            s0 = wmma_bf16(qa[kk], b0, s0);
            s1 = wmma_bf16(qa[kk], b1, s1);
        }
        if (kt + 1 <= ktend)
            __builtin_prefetch(Kp + (size_t)(kb + 32) * DK, 0, 3);

#pragma unroll
        for (int r = 0; r < 8; ++r) {
            const int q  = q0 + r + half * 8;
            const int ky = kb + (lane & 15);
            float v0 = (ky      <= q) ? s0[r] * SCALE : -1e30f;
            float v1 = (ky + 16 <= q) ? s1[r] * SCALE : -1e30f;
            float rmax = rowmax16(fmaxf(v0, v1));
            float mnew = fmaxf(mrow[r], rmax);
            float alpha = __expf(mrow[r] - mnew);
            float p0 = __expf(v0 - mnew);
            float p1 = __expf(v1 - mnew);
            lrow[r] = lrow[r] * alpha + rowsum16(p0 + p1);
            mrow[r] = mnew;
#pragma unroll
            for (int nb = 0; nb < 8; ++nb) o[nb][r] *= alpha;
            const int row = r + half * 8;
            smP[w][row * 32 + (lane & 15)]      = (__bf16)p0;
            smP[w][row * 32 + (lane & 15) + 16] = (__bf16)p1;
        }
        asm volatile("s_wait_dscnt 0" ::: "memory");   // P stores -> P loads (same wave)
        v16bf pf = load_a_frag(&smP[w][0], 32);
#pragma unroll
        for (int nb = 0; nb < 8; ++nb) {
            v16bf vb = load_b_frag(Vp + (size_t)(nb * 16) * S_ + kb, S_);
            o[nb] = wmma_bf16(pf, vb, o[nb]);
        }
    }

    const int b = bh >> 4, h = bh & 15;
#pragma unroll
    for (int nb = 0; nb < 8; ++nb) {
#pragma unroll
        for (int r = 0; r < 8; ++r) {
            const int q  = q0 + r + half * 8;
            const int vd = nb * 16 + (lane & 15);
            O[((size_t)(b * S_) + q) * D_ + h * DK + vd] = (__bf16)(o[nb][r] / lrow[r]);
        }
    }
}

// ------------------------------------------------------- output projection

__global__ __launch_bounds__(THREADS)
void gemm_out_kernel(const __bf16* __restrict__ O,
                     const __bf16* __restrict__ W,
                     float* __restrict__ out) {
    __shared__ __bf16 smA[2 * TILE_ELEMS];
    v8f acc[4][2];
#pragma unroll
    for (int i = 0; i < 4; ++i)
#pragma unroll
        for (int j = 0; j < 2; ++j) acc[i][j] = zero8();

    gemm_mainloop(O, W, D_, BS, smA, acc);

    const int lane = threadIdx.x & 31;
    const int wave = threadIdx.x >> 5;
    const int wm = wave & 1, wn = wave >> 1;
    const int half = (lane >> 4) & 1;
#pragma unroll
    for (int i = 0; i < 4; ++i) {
#pragma unroll
        for (int j = 0; j < 2; ++j) {
            const int n = blockIdx.y * 128 + wn * 32 + j * 16 + (lane & 15);
            const int mbase = blockIdx.x * 128 + wm * 64 + i * 16 + half * 8;
#pragma unroll
            for (int r = 0; r < 8; ++r)
                out[(size_t)(mbase + r) * D_ + n] = acc[i][j][r];
        }
    }
}

// ---------------------------------------------------------------- launch

extern "C" void kernel_launch(void* const* d_in, const int* in_sizes, int n_in,
                              void* d_out, int out_size, void* d_ws, size_t ws_size,
                              hipStream_t stream) {
    const float* x  = (const float*)d_in[0];
    const float* wq = (const float*)d_in[1];
    const float* wk = (const float*)d_in[2];
    const float* wv = (const float*)d_in[3];
    const float* wo = (const float*)d_in[4];
    float* out = (float*)d_out;

    // Workspace carve-up (bf16 buffers), ~160 MB total. O reuses xb region.
    char* ws = (char*)d_ws;
    size_t off = 0;
    __bf16* xb   = (__bf16*)(ws + off); off += (size_t)BS * D_ * 2;     // 32MB (also O)
    __bf16* wqkv = (__bf16*)(ws + off); off += (size_t)NQKV * D_ * 2;   // 24MB
    __bf16* wob  = (__bf16*)(ws + off); off += (size_t)D_ * D_ * 2;     //  8MB
    __bf16* Qb   = (__bf16*)(ws + off); off += (size_t)BS * D_ * 2;     // 32MB
    __bf16* Kb   = (__bf16*)(ws + off); off += (size_t)BS * D_ * 2;     // 32MB
    __bf16* Vt   = (__bf16*)(ws + off); off += (size_t)BS * D_ * 2;     // 32MB
    __bf16* Ob   = xb;                                                  // reuse

    const int nX = BS * D_, nW = D_ * D_;
    auto blocks = [](int n) { return (n + THREADS * 4 - 1) / (THREADS * 4); };
    cvt_f32_bf16_kernel<<<blocks(nX), THREADS, 0, stream>>>(x, xb, nX);
    cvt_f32_bf16_kernel<<<blocks(nW), THREADS, 0, stream>>>(wq, wqkv,            nW);
    cvt_f32_bf16_kernel<<<blocks(nW), THREADS, 0, stream>>>(wk, wqkv + (size_t)nW,     nW);
    cvt_f32_bf16_kernel<<<blocks(nW), THREADS, 0, stream>>>(wv, wqkv + (size_t)2 * nW, nW);
    cvt_f32_bf16_kernel<<<blocks(nW), THREADS, 0, stream>>>(wo, wob, nW);

    dim3 gQKV(BS / 128, NQKV / 128);          // 64 x 48
    gemm_qkv_rope_kernel<<<gQKV, THREADS, 0, stream>>>(xb, wqkv, Qb, Kb, Vt);

    flash_attn_kernel<<<B_ * H_ * (S_ / 128), THREADS, 0, stream>>>(Qb, Kb, Vt, Ob);

    dim3 gOut(BS / 128, D_ / 128);            // 64 x 16
    gemm_out_kernel<<<gOut, THREADS, 0, stream>>>(Ob, wob, out);
}
